// AttentionLayer_62285615727309
// MI455X (gfx1250) — compile-verified, matrix operand order checked by reading, not run on hardware
//
#include <hip/hip_runtime.h>

// ---------------------------------------------------------------------------
// MI455X (gfx1250) DIN-style attention, bf16 WMMA + async global->LDS staging.
//   x = [k, q, k-q, q*k] @ W1 -> sigmoid -> @W2 -> sigmoid -> @W3 -> sigmoid
//   masked softmax over T, weighted sum of keys.
// Restructured: x@W1 = k@(W1a+W1c) + (q*k)@W1d + perBatchBias(q@(W1b-W1c))
// M is processed in 2 chunks (112 + 96 rows) so LDS stays under 160 KB
// -> 2 workgroups per 320KB WGP for barrier/latency overlap.
// Sigmoid epilogue uses v_exp_f32 + v_rcp_f32 (no IEEE div expansion).
// ---------------------------------------------------------------------------

typedef __bf16 v16bf __attribute__((ext_vector_type(16)));
typedef __bf16 v8bf  __attribute__((ext_vector_type(8)));
typedef float  v8f   __attribute__((ext_vector_type(8)));

union AV { v16bf v; v8bf h[2]; };

#define Tlen   200
#define TP     208      // T padded to 13 * 16
#define Ddim   128
#define H1dim  256
#define H2dim  128

// Dynamic LDS layout (byte offsets; no static LDS => dynamic base == 0):
//   [0,      57344) R1: Ash bf16 [112][256]  (later aliased by H2 [112][128])
//   [57344, 114688) R2: Kst f32  [112][128] staging (aliased by H1 [112][256])
//   [114688,115200) qsh   f32[128]
//   [115200,116224) bias1 f32[256]
//   [116224,117248) sc    f32[208]+pad
//   [117248,118272) red   f32[256]
#define R2_OFF     57344
#define SMEM_BYTES 118272

// sigmoid via native exp + single-op v_rcp_f32 (~1 ulp; GEMM epilogue hot path)
__device__ __forceinline__ float sigmoidf(float x) {
  return __builtin_amdgcn_rcpf(1.0f + __expf(-x));
}

// Async DMA: 16 bytes per lane, global -> LDS, tracked by ASYNCcnt.
__device__ __forceinline__ void async_b128_to_lds(unsigned lds_byte_off,
                                                  const float* gaddr) {
  asm volatile("global_load_async_to_lds_b128 %0, %1, off"
               :: "v"(lds_byte_off), "v"(gaddr) : "memory");
}
__device__ __forceinline__ void wait_asynccnt0() {
  asm volatile("s_wait_asynccnt 0x0" ::: "memory");
}

// Stage validRows*128 f32 keys into LDS R2 via async B128 copies.
__device__ __forceinline__ void issue_keys_async(const float* __restrict__ kb,
                                                 int row0, int validRows, int tid) {
  const float* base = kb + (size_t)row0 * Ddim;
  for (int i = tid; i < validRows * 32; i += 256)          // 32 x b128 per row
    async_b128_to_lds((unsigned)(R2_OFF + i * 16), base + i * 4);
}

// One 16x16 output tile, K = 8*32 = 256, bf16 WMMA.
// arow/bcol pre-offset by lane: &X[(tile*16 + lane%16)*256 + (lane/16)*8]
// (16-bit fragment layout, ISA 7.12.2: lanes 0-15 hold K 0-7/16-23,
//  lanes 16-31 hold K 8-15/24-31).
__device__ __forceinline__ v8f mma_k256(const __bf16* __restrict__ arow,
                                        const __bf16* __restrict__ bcol) {
  v8f c = {};
#pragma unroll
  for (int kt = 0; kt < 8; ++kt) {
    AV a, b;
    a.h[0] = *(const v8bf*)(arow + kt * 32);
    a.h[1] = *(const v8bf*)(arow + kt * 32 + 16);
    b.h[0] = *(const v8bf*)(bcol + kt * 32);
    b.h[1] = *(const v8bf*)(bcol + kt * 32 + 16);
    c = __builtin_amdgcn_wmma_f32_16x16x32_bf16(false, a.v, false, b.v,
                                                (short)0, c, false, false);
  }
  return c;
}

// ---------------------------------------------------------------------------
// Prep kernel: fused/transposed bf16 weights in workspace.
//   W1f_t[n*256+k] = bf16( W1[(k+256)*256+n] + (k<128 ? W1[k*256+n] : 0) )
//   W2t  [n*256+k] = bf16( W2[k*128+n] )
//   W1bc [d*256+h] = W1[(128+d)*256+h] - W1[(256+d)*256+h]
// ---------------------------------------------------------------------------
__global__ __launch_bounds__(256)
void attn_prep(const float* __restrict__ W1, const float* __restrict__ W2,
               __bf16* __restrict__ W1f_t, __bf16* __restrict__ W2t,
               float* __restrict__ W1bc) {
  int idx = blockIdx.x * 256 + threadIdx.x;
  if (idx < 256 * 256) {
    int n = idx >> 8, k = idx & 255;
    float v = W1[(k + 256) * 256 + n];
    if (k < 128) v += W1[k * 256 + n];
    W1f_t[n * 256 + k] = (__bf16)v;
  } else if (idx < 256 * 256 + 128 * 256) {
    int j = idx - 256 * 256;
    int n = j >> 8, k = j & 255;
    W2t[n * 256 + k] = (__bf16)W2[k * 128 + n];
  } else if (idx < 256 * 256 + 2 * 128 * 256) {
    int j = idx - 256 * 256 - 128 * 256;
    int d = j >> 8, h = j & 255;
    W1bc[d * 256 + h] = W1[(128 + d) * 256 + h] - W1[(256 + d) * 256 + h];
  }
}

// ---------------------------------------------------------------------------
// Per-chunk pipeline: convert staged keys -> GEMM1 -> GEMM2 -> scores.
// ROWS: padded chunk rows (mult of 16); VALID: rows with real data.
// ---------------------------------------------------------------------------
template <int ROWS, int VALID, bool PREFETCH_NEXT>
__device__ __forceinline__ void process_chunk(
    int row0, const float* __restrict__ kb, const int* __restrict__ mask_b,
    const __bf16* __restrict__ W1f_t, const __bf16* __restrict__ W2t,
    const float* __restrict__ b2, const float* __restrict__ W3, float b3v,
    __bf16* Ash, float* Kst, __bf16* H1sh, __bf16* H2sh,
    const float* qsh, const float* bias1, float* sc,
    int tid, int wave, int lm, int lh) {
  constexpr int NT_M = ROWS / 16;

  // ---- convert staged f32 keys -> A = [k | q*k] bf16 ------------------------
  for (int i = tid; i < ROWS * Ddim; i += 256) {
    int t = i >> 7, d = i & 127;
    float kv = (t < VALID) ? Kst[i] : 0.0f;
    Ash[t * 256 + d]       = (__bf16)kv;
    Ash[t * 256 + 128 + d] = (__bf16)(kv * qsh[d]);
  }
  __syncthreads();   // Ash ready; Kst dead (H1 may overwrite)

  // ---- GEMM1: H1 = sigmoid(A[ROWS,256] @ W1f[256,256] + bias1) --------------
  for (int tix = wave; tix < NT_M * 16; tix += 8) {
    int mt = tix % NT_M, nt = tix / NT_M;
    const __bf16* arow = &Ash[(mt * 16 + lm) * 256 + lh * 8];
    const __bf16* bcol = &W1f_t[(nt * 16 + lm) * 256 + lh * 8];
    v8f c = mma_k256(arow, bcol);
    int n = nt * 16 + lm;
    float bn = bias1[n];
#pragma unroll
    for (int r = 0; r < 8; ++r) {
      int m = mt * 16 + lh * 8 + r;
      H1sh[m * 256 + n] = (__bf16)sigmoidf(c[r] + bn);
    }
  }
  __syncthreads();

  // ---- GEMM2: H2 = sigmoid(H1[ROWS,256] @ W2[256,128] + b2) -----------------
  for (int tix = wave; tix < NT_M * 8; tix += 8) {
    int mt = tix % NT_M, nt = tix / NT_M;
    const __bf16* arow = &H1sh[(mt * 16 + lm) * 256 + lh * 8];
    const __bf16* bcol = &W2t[(nt * 16 + lm) * 256 + lh * 8];
    v8f c = mma_k256(arow, bcol);
    int n = nt * 16 + lm;
    float bn = b2[n];
#pragma unroll
    for (int r = 0; r < 8; ++r) {
      int m = mt * 16 + lh * 8 + r;
      H2sh[m * 128 + n] = (__bf16)sigmoidf(c[r] + bn);
    }
  }
  __syncthreads();   // H1/Kst region (R2) now dead

  // ---- overlap: start DMA of the next chunk's keys into R2 ------------------
  if (PREFETCH_NEXT)
    issue_keys_async(kb, 112, 88, tid);   // chunk1: rows 112..199

  // ---- scores: sc[t] = mask ? sigmoid(H2 @ W3 + b3) : -inf ------------------
  if (tid < ROWS) {
    float v = -__builtin_inff();
    if (tid < VALID) {
      float acc = 0.0f;
#pragma unroll 4
      for (int k = 0; k < H2dim; ++k)
        acc += (float)H2sh[tid * 128 + k] * W3[k];
      float s = sigmoidf(acc + b3v);
      v = (mask_b[row0 + tid] == 1) ? s : -__builtin_inff();
    }
    sc[row0 + tid] = v;
  }
  if (PREFETCH_NEXT) wait_asynccnt0();
  __syncthreads();
}

// ---------------------------------------------------------------------------
// Main kernel: one workgroup (8 wave32) per batch row; ~118 KB LDS -> 2 WG/WGP.
// ---------------------------------------------------------------------------
__global__ __launch_bounds__(256)
void attn_main(const float* __restrict__ query, const float* __restrict__ keys,
               const int* __restrict__ mask,
               const float* __restrict__ b1, const float* __restrict__ b2,
               const float* __restrict__ W3, const float* __restrict__ b3,
               const __bf16* __restrict__ W1f_t, const __bf16* __restrict__ W2t,
               const float* __restrict__ W1bc,
               float* __restrict__ out) {
  extern __shared__ char smem[];
  __bf16* Ash   = (__bf16*)(smem);
  float*  Kst   = (float*)(smem + R2_OFF);
  __bf16* H1sh  = (__bf16*)(smem + R2_OFF);
  float*  qsh   = (float*)(smem + 114688);
  float*  bias1 = (float*)(smem + 115200);
  float*  sc    = (float*)(smem + 116224);
  float*  red   = (float*)(smem + 117248);
  __bf16* H2sh  = Ash;

  const int tid  = threadIdx.x;
  const int b    = blockIdx.x;
  const int wave = tid >> 5;
  const int lane = tid & 31;
  const int lm   = lane & 15;   // n (B/C) or m (A) within tile
  const int lh   = lane >> 4;   // K-half selector

  const float* kb      = keys + (size_t)b * Tlen * Ddim;
  const int*   mask_b  = mask + (size_t)b * Tlen;

  // ---- kick off async DMA of chunk-0 keys while we do the bias fold ---------
  issue_keys_async(kb, 0, 112, tid);

  if (tid < Ddim) qsh[tid] = query[(size_t)b * Ddim + tid];
  __syncthreads();

  // per-batch bias fold: bias1 = b1 + q @ (W1b - W1c)
  {
    float acc = b1[tid];
#pragma unroll 4
    for (int d = 0; d < Ddim; ++d) acc += qsh[d] * W1bc[d * 256 + tid];
    bias1[tid] = acc;
  }
  wait_asynccnt0();
  __syncthreads();

  float b3v = b3[0];
  process_chunk<112, 112, true >(0,   kb, mask_b, W1f_t, W2t, b2, W3, b3v,
                                 Ash, Kst, H1sh, H2sh, qsh, bias1, sc,
                                 tid, wave, lm, lh);
  process_chunk< 96,  88, false>(112, kb, mask_b, W1f_t, W2t, b2, W3, b3v,
                                 Ash, Kst, H1sh, H2sh, qsh, bias1, sc,
                                 tid, wave, lm, lh);

  // ---- softmax over T -------------------------------------------------------
  red[tid] = (tid < TP) ? sc[tid] : -__builtin_inff();
  __syncthreads();
  for (int s = 128; s > 0; s >>= 1) {
    if (tid < s) red[tid] = fmaxf(red[tid], red[tid + s]);
    __syncthreads();
  }
  float mx = red[0];
  __syncthreads();
  float e = 0.0f;
  if (tid < Tlen) e = __expf(sc[tid] - mx);
  red[tid] = e;
  __syncthreads();
  for (int s = 128; s > 0; s >>= 1) {
    if (tid < s) red[tid] += red[tid + s];
    __syncthreads();
  }
  float inv = __builtin_amdgcn_rcpf(red[0]);
  __syncthreads();
  if (tid < Tlen) sc[tid] = e * inv;
  __syncthreads();

  // ---- out[b,:] = attn @ keys (keys re-read; L2-resident) -------------------
  {
    int d = tid & 127, half = tid >> 7;
    float acc = 0.0f;
    const float* kp = kb + (size_t)(half * 100) * Ddim + d;
#pragma unroll 4
    for (int t = 0; t < 100; ++t) acc += sc[half * 100 + t] * kp[t * Ddim];
    red[tid] = acc;
    __syncthreads();
    if (tid < Ddim) out[(size_t)b * Ddim + tid] = red[tid] + red[tid + 128];
  }
}

// ---------------------------------------------------------------------------
extern "C" void kernel_launch(void* const* d_in, const int* in_sizes, int n_in,
                              void* d_out, int out_size, void* d_ws, size_t ws_size,
                              hipStream_t stream) {
  const float* query = (const float*)d_in[0];
  const float* keys  = (const float*)d_in[1];
  const int*   mask  = (const int*)d_in[2];
  const float* W1    = (const float*)d_in[3];
  const float* b1    = (const float*)d_in[4];
  const float* W2    = (const float*)d_in[5];
  const float* b2    = (const float*)d_in[6];
  const float* W3    = (const float*)d_in[7];
  const float* b3    = (const float*)d_in[8];
  float* out = (float*)d_out;

  // Workspace: fused/transposed bf16 weights + f32 bias-fold matrix (320 KB).
  char* ws = (char*)d_ws;
  __bf16* W1f_t = (__bf16*)(ws);            // 256*256*2 = 131072 B
  __bf16* W2t   = (__bf16*)(ws + 131072);   // 128*256*2 =  65536 B
  float*  W1bc  = (float*)(ws + 196608);    // 128*256*4 = 131072 B

  int B = in_sizes[0] / Ddim;

  attn_prep<<<512, 256, 0, stream>>>(W1, W2, W1f_t, W2t, W1bc);

  (void)hipFuncSetAttribute((const void*)attn_main,
                            hipFuncAttributeMaxDynamicSharedMemorySize,
                            SMEM_BYTES);
  attn_main<<<B, 256, SMEM_BYTES, stream>>>(query, keys, mask, b1, b2, W3, b3,
                                            W1f_t, W2t, W1bc, out);
}
